// RoPECausalAttention_33054068310127
// MI455X (gfx1250) — compile-verified, matrix operand order checked by reading, not run on hardware
//
#include <hip/hip_runtime.h>

#define DEVI __device__ __forceinline__

typedef __attribute__((ext_vector_type(16))) __bf16 bf16x16;
typedef __attribute__((ext_vector_type(8)))  float  floatx8;
typedef __attribute__((ext_vector_type(4)))  int    int4v;

typedef __attribute__((address_space(1))) int4v as1_int4;
typedef __attribute__((address_space(3))) int4v as3_int4;

#if defined(__has_builtin)
#if __has_builtin(__builtin_amdgcn_global_load_async_to_lds_b128) && \
    __has_builtin(__builtin_amdgcn_s_wait_asynccnt)
#define USE_ASYNC_LDS 1
#endif
#endif
#ifndef USE_ASYNC_LDS
#define USE_ASYNC_LDS 0
#endif

namespace {
constexpr int kB = 2, kC = 2048, kD = 1024, kH = 16, kDH = 64;
}

// 16-byte global -> LDS copy. On gfx1250 toolchains with the async builtins this
// lowers to global_load_async_to_lds_b128 (ASYNCcnt); otherwise a plain
// load/ds_store pair.
DEVI void copy16B_to_lds(__bf16* lds, const __bf16* g) {
#if USE_ASYNC_LDS
  __builtin_amdgcn_global_load_async_to_lds_b128((as1_int4*)g, (as3_int4*)lds, 0, 0);
#else
  *(uint4*)lds = *(const uint4*)g;
#endif
}

// Drain this wave's outstanding async copies, then block-wide barrier.
DEVI void stage_join() {
#if USE_ASYNC_LDS
  __builtin_amdgcn_s_wait_asynccnt(0);
#endif
  __syncthreads();
}

// D = A(16x32) * B(32x16) + C, bf16 in / f32 accumulate (CDNA5 WMMA)
DEVI floatx8 wmma_bf16(bf16x16 a, bf16x16 b, floatx8 c) {
  return __builtin_amdgcn_wmma_f32_16x16x32_bf16(
      /*neg_a=*/false, a, /*neg_b=*/false, b,
      /*c_mod=*/(short)0, c, /*reuse_a=*/false, /*reuse_b=*/false);
}

// Load a 16x32 bf16 A- or B-fragment from a row-major tile (row = M for A,
// row = N for B, contiguous contraction dim). CDNA5 16-bit operand layout:
// lane L -> row L&15, K = {kb..kb+7, kb+16..kb+23}, kb = 8*(L>=16).
DEVI bf16x16 load_frag16x32(const __bf16* tile, int ld) {
  const int lane = threadIdx.x & 31;
  const __bf16* p = tile + (size_t)(lane & 15) * ld + ((lane >> 4) << 3);
  union { bf16x16 v; uint4 q[2]; } f;
  f.q[0] = *(const uint4*)(p);
  f.q[1] = *(const uint4*)(p + 16);
  return f.v;
}

__global__ __launch_bounds__(256) void cvt_f32_to_bf16(
    const float* __restrict__ src, __bf16* __restrict__ dst, int n4) {
  int i = blockIdx.x * blockDim.x + threadIdx.x;
  if (i < n4) {
    float4 f = ((const float4*)src)[i];
    dst[4 * i + 0] = (__bf16)f.x;
    dst[4 * i + 1] = (__bf16)f.y;
    dst[4 * i + 2] = (__bf16)f.z;
    dst[4 * i + 3] = (__bf16)f.w;
  }
}

// dst(N x K, bf16) = transpose(src(K x N, f32)). One-shot preprocessing so the
// GEMM B operand is n-major with contiguous K (pure b128 staging, no scatter).
__global__ __launch_bounds__(256) void transpose_f32_to_bf16(
    const float* __restrict__ src, __bf16* __restrict__ dst, int K, int N) {
  const int k = blockIdx.x * blockDim.x + threadIdx.x;
  const int n = blockIdx.y;
  dst[(size_t)n * K + k] = (__bf16)src[(size_t)k * N + n];
}

// C(MxN, f32) = A(MxK, bf16 row-major) * Bt(NxK, bf16 row-major)^T [+ bias]
// Block tile 128x128, K-chunks of 32, double-buffered async staging.
// 8 waves: 4 along M (32 rows, 2 A-frags), 2 along N (64 cols, 4 B-frags).
__global__ __launch_bounds__(256) void gemm_bf16_wmma(
    const __bf16* __restrict__ A, const __bf16* __restrict__ Bt,
    const float* __restrict__ bias, float* __restrict__ Cm,
    int N, int K) {
  __shared__ __bf16 sA[2][128][32];
  __shared__ __bf16 sB[2][128][32];
  const int t    = threadIdx.x;
  const int wave = t >> 5;
  const int lane = t & 31;
  const int bm = blockIdx.y * 128;
  const int bn = blockIdx.x * 128;
  const int m0 = (wave >> 1) * 32;
  const int n0 = (wave & 1) * 64;
  floatx8 acc[2][4] = {};

  const int row = t >> 1;            // 0..127
  const int col = (t & 1) * 16;      // 0 or 16
  auto stage = [&](int buf, int k0) {
    const __bf16* sa = A  + (size_t)(bm + row) * K + k0 + col;
    const __bf16* sb = Bt + (size_t)(bn + row) * K + k0 + col;
    copy16B_to_lds(&sA[buf][row][col],     sa);
    copy16B_to_lds(&sA[buf][row][col + 8], sa + 8);
    copy16B_to_lds(&sB[buf][row][col],     sb);
    copy16B_to_lds(&sB[buf][row][col + 8], sb + 8);
  };

  const int nIter = K / 32;
  stage(0, 0);
  for (int i = 0; i < nIter; ++i) {
    stage_join();                                   // buffer i ready block-wide
    if (i + 1 < nIter) stage((i + 1) & 1, (i + 1) * 32);
    const __bf16(*cA)[32] = sA[i & 1];
    const __bf16(*cB)[32] = sB[i & 1];
    bf16x16 a0 = load_frag16x32(&cA[m0][0], 32);
    bf16x16 a1 = load_frag16x32(&cA[m0 + 16][0], 32);
#pragma unroll
    for (int nc = 0; nc < 4; ++nc) {
      bf16x16 b = load_frag16x32(&cB[n0 + nc * 16][0], 32);
      acc[0][nc] = wmma_bf16(a0, b, acc[0][nc]);
      acc[1][nc] = wmma_bf16(a1, b, acc[1][nc]);
    }
  }

  const int nl = lane & 15;
  const int mh = (lane >> 4) << 3;  // C-frag: element j -> row mh+j, col nl
#pragma unroll
  for (int nc = 0; nc < 4; ++nc) {
    const int c = bn + n0 + nc * 16 + nl;
    const float bv = bias ? bias[c] : 0.0f;
#pragma unroll
    for (int mi = 0; mi < 2; ++mi)
#pragma unroll
      for (int j = 0; j < 8; ++j)
        Cm[(size_t)(bm + m0 + mi * 16 + mh + j) * N + c] = acc[mi][nc][j] + bv;
  }
}

// qkv(B,C,3D f32) -> bf16 q,k,q_rot,k_rot in (B,H,C,dh) and V transposed to
// (B,H,dh,C) so attention V-staging is contiguous. Reference theta quirk kept.
__global__ __launch_bounds__(256) void rope_pack_kernel(
    const float* __restrict__ qkv, __bf16* __restrict__ qn,
    __bf16* __restrict__ kn, __bf16* __restrict__ vt,
    __bf16* __restrict__ qr, __bf16* __restrict__ kr) {
  const int idx = blockIdx.x * blockDim.x + threadIdx.x;  // (b, c, h)
  const int h = idx & (kH - 1);
  const int c = (idx >> 4) & (kC - 1);
  const int b = idx >> 15;
  const float* rowp = qkv + (size_t)(b * kC + c) * (3 * kD);
  const float* qp = rowp + h * kDH;
  const float* kp = rowp + kD + h * kDH;
  const float* vp = rowp + 2 * kD + h * kDH;
  const size_t ob  = ((size_t)(b * kH + h) * kC + c) * kDH;
  const size_t vtb = (size_t)(b * kH + h) * kDH * kC + c;   // + d*kC
  const float l2t = 13.28771237954945f;  // log2(10000)
#pragma unroll 4
  for (int i = 0; i < kDH / 2; ++i) {
    float theta = exp2f(-2.0f * ((float)i - 1.0f) / (float)kDH * l2t);
    float sn, cs;
    __sincosf((float)c * theta, &sn, &cs);
    float qe = qp[2 * i], qo = qp[2 * i + 1];
    float ke = kp[2 * i], ko = kp[2 * i + 1];
    qr[ob + 2 * i]     = (__bf16)(qe * cs - qo * sn);
    qr[ob + 2 * i + 1] = (__bf16)(qe * sn + qo * cs);
    kr[ob + 2 * i]     = (__bf16)(ke * cs - ko * sn);
    kr[ob + 2 * i + 1] = (__bf16)(ke * sn + ko * cs);
    qn[ob + 2 * i]     = (__bf16)qe;
    qn[ob + 2 * i + 1] = (__bf16)qo;
    kn[ob + 2 * i]     = (__bf16)ke;
    kn[ob + 2 * i + 1] = (__bf16)ko;
    vt[vtb + (size_t)(2 * i) * kC]     = (__bf16)vp[2 * i];
    vt[vtb + (size_t)(2 * i + 1) * kC] = (__bf16)vp[2 * i + 1];
  }
}

// One (b,h) per blockIdx.y, 64 q rows per block (4 waves x 16 q rows).
// Double-buffered 32-key tiles via async LDS copies. Denominator sums
// exp(q.k*scale) over ALL keys (reference is unmasked); numerator path uses
// rotated q/k with causal mask, P round-trips through LDS (C-frag -> A-frag),
// then P @ V via WMMA, divide at the end.
__global__ __launch_bounds__(128) void attn_kernel(
    const __bf16* __restrict__ qn, const __bf16* __restrict__ kn,
    const __bf16* __restrict__ qr, const __bf16* __restrict__ kr,
    const __bf16* __restrict__ vt, __bf16* __restrict__ yb) {
  __shared__ __bf16 sKr[2][32][64];   // keys x dh (contiguous d)
  __shared__ __bf16 sKn[2][32][64];
  __shared__ __bf16 sVt[2][64][32];   // dh x keys (contiguous k) from vt
  __shared__ __bf16 sP[4][16][32];    // per-wave P tile, m x keys

  const int t = threadIdx.x;
  const int wave = t >> 5;
  const int lane = t & 31;
  const int bh = blockIdx.y;               // b*H + h
  const int qw = blockIdx.x * 64 + wave * 16;
  const size_t base  = (size_t)bh * kC * kDH;
  const size_t vbase = (size_t)bh * kDH * kC;
  const float scale = 0.125f;              // 1/sqrt(64)

  const size_t qoff = base + (size_t)qw * kDH;
  bf16x16 aqr0 = load_frag16x32(qr + qoff, kDH);
  bf16x16 aqr1 = load_frag16x32(qr + qoff + 32, kDH);
  bf16x16 aqn0 = load_frag16x32(qn + qoff, kDH);
  bf16x16 aqn1 = load_frag16x32(qn + qoff + 32, kDH);

  floatx8 yacc[4] = {};
  float denp[8] = {0.f, 0.f, 0.f, 0.f, 0.f, 0.f, 0.f, 0.f};
  const int nl = lane & 15;
  const int mh = (lane >> 4) << 3;
  const int qhi = qw + 15;

  const int krow = t >> 2;             // 0..31 (keys)
  const int kcol = (t & 3) * 16;       // d chunk
  const int vrow = t >> 1;             // 0..63 (dh)
  const int vcol = (t & 1) * 16;       // key chunk
  auto stage = [&](int buf, int k0) {
    const __bf16* sr = kr + base + (size_t)(k0 + krow) * kDH + kcol;
    const __bf16* sn = kn + base + (size_t)(k0 + krow) * kDH + kcol;
    const __bf16* sv = vt + vbase + (size_t)vrow * kC + k0 + vcol;
    copy16B_to_lds(&sKr[buf][krow][kcol],     sr);
    copy16B_to_lds(&sKr[buf][krow][kcol + 8], sr + 8);
    copy16B_to_lds(&sKn[buf][krow][kcol],     sn);
    copy16B_to_lds(&sKn[buf][krow][kcol + 8], sn + 8);
    copy16B_to_lds(&sVt[buf][vrow][vcol],     sv);
    copy16B_to_lds(&sVt[buf][vrow][vcol + 8], sv + 8);
  };

  const int nIter = kC / 32;
  stage(0, 0);
  for (int i = 0; i < nIter; ++i) {
    stage_join();
    if (i + 1 < nIter) stage((i + 1) & 1, (i + 1) * 32);
    const int k0 = i * 32;
    const __bf16(*cKr)[64] = sKr[i & 1];
    const __bf16(*cKn)[64] = sKn[i & 1];
    const __bf16(*cVt)[32] = sVt[i & 1];

    // denominator: plain scores over every key tile, no mask
#pragma unroll
    for (int sub = 0; sub < 2; ++sub) {
      bf16x16 b0 = load_frag16x32(&cKn[sub * 16][0], 64);
      bf16x16 b1 = load_frag16x32(&cKn[sub * 16][32], 64);
      floatx8 s = {};
      s = wmma_bf16(aqn0, b0, s);
      s = wmma_bf16(aqn1, b1, s);
#pragma unroll
      for (int j = 0; j < 8; ++j) denp[j] += __expf(s[j] * scale);
    }

    // numerator + P@V only for causal tiles (wave-uniform branch, EXEC stays full)
    if (k0 <= qhi) {
#pragma unroll
      for (int sub = 0; sub < 2; ++sub) {
        bf16x16 b0 = load_frag16x32(&cKr[sub * 16][0], 64);
        bf16x16 b1 = load_frag16x32(&cKr[sub * 16][32], 64);
        floatx8 s = {};
        s = wmma_bf16(aqr0, b0, s);
        s = wmma_bf16(aqr1, b1, s);
        const int kg = k0 + sub * 16 + nl;
#pragma unroll
        for (int j = 0; j < 8; ++j) {
          const int qg = qw + mh + j;
          float p = (kg <= qg) ? __expf(s[j] * scale) : 0.0f;
          sP[wave][mh + j][sub * 16 + nl] = (__bf16)p;
        }
      }
      bf16x16 ap = load_frag16x32(&sP[wave][0][0], 32);
#pragma unroll
      for (int nc = 0; nc < 4; ++nc) {
        bf16x16 bv = load_frag16x32(&cVt[nc * 16][0], 32);
        yacc[nc] = wmma_bf16(ap, bv, yacc[nc]);
      }
    }
  }

  // finish denominator: sum 16 lanes within each half (C-frag row groups)
#pragma unroll
  for (int j = 0; j < 8; ++j) {
    float d = denp[j];
    d += __shfl_xor(d, 1, 32);
    d += __shfl_xor(d, 2, 32);
    d += __shfl_xor(d, 4, 32);
    d += __shfl_xor(d, 8, 32);
    denp[j] = d;
  }

  const int b = bh >> 4;
  const int h = bh & (kH - 1);
#pragma unroll
  for (int nc = 0; nc < 4; ++nc) {
#pragma unroll
    for (int j = 0; j < 8; ++j) {
      const int tok = b * kC + qw + mh + j;
      const int c = h * kDH + nc * 16 + nl;
      yb[(size_t)tok * kD + c] = (__bf16)(yacc[nc][j] / denp[j]);
    }
  }
}

extern "C" void kernel_launch(void* const* d_in, const int* in_sizes, int n_in,
                              void* d_out, int out_size, void* d_ws, size_t ws_size,
                              hipStream_t stream) {
  const float* x    = (const float*)d_in[0];
  const float* Wqkv = (const float*)d_in[1];
  const float* Wout = (const float*)d_in[2];
  const float* bout = (const float*)d_in[3];
  float* out = (float*)d_out;

  const size_t szX    = (size_t)kB * kC * kD;   // 4,194,304
  const size_t szWqkv = (size_t)kD * 3 * kD;
  const size_t szWout = (size_t)kD * kD;

  char* w = (char*)d_ws;
  auto carve = [&](size_t bytes) -> void* {
    void* p = (void*)w;
    w += (bytes + 255) & ~(size_t)255;
    return p;
  };
  __bf16* xb     = (__bf16*)carve(szX * 2);
  __bf16* wqkvT  = (__bf16*)carve(szWqkv * 2);   // (3D x D)
  __bf16* woutT  = (__bf16*)carve(szWout * 2);   // (D x D)
  float*  qkv    = (float*)carve((size_t)kB * kC * 3 * kD * 4);
  __bf16* qn     = (__bf16*)carve(szX * 2);
  __bf16* kn     = (__bf16*)carve(szX * 2);
  __bf16* vt     = (__bf16*)carve(szX * 2);
  __bf16* qr     = (__bf16*)carve(szX * 2);
  __bf16* kr     = (__bf16*)carve(szX * 2);
  __bf16* yb     = (__bf16*)carve(szX * 2);

  // 1) operand precision/layout prep
  cvt_f32_to_bf16<<<(int)(szX / 4 / 256), 256, 0, stream>>>(x, xb, (int)(szX / 4));
  transpose_f32_to_bf16<<<dim3(kD / 256, 3 * kD), 256, 0, stream>>>(Wqkv, wqkvT, kD, 3 * kD);
  transpose_f32_to_bf16<<<dim3(kD / 256, kD), 256, 0, stream>>>(Wout, woutT, kD, kD);

  // 2) qkv = x @ Wqkv   (4096x1024 @ 1024x3072)
  gemm_bf16_wmma<<<dim3(3 * kD / 128, kB * kC / 128), 256, 0, stream>>>(
      xb, wqkvT, nullptr, qkv, 3 * kD, kD);

  // 3) RoPE + pack (q/k row-major per head, V transposed per head)
  rope_pack_kernel<<<(kB * kC * kH) / 256, 256, 0, stream>>>(qkv, qn, kn, vt, qr, kr);

  // 4) fused attention -> yb (B*C, D) bf16
  attn_kernel<<<dim3(kC / 64, kB * kH), 128, 0, stream>>>(qn, kn, qr, kr, vt, yb);

  // 5) out = y @ Wout + bout   (4096x1024 @ 1024x1024)
  gemm_bf16_wmma<<<dim3(kD / 128, kB * kC / 128), 256, 0, stream>>>(
      yb, woutT, bout, out, kD, kD);

  (void)in_sizes; (void)n_in; (void)out_size; (void)ws_size;
}
